// QuantumNATEnhanced_65481071399515
// MI455X (gfx1250) — matrix operand order repaired and checked by reading, not compiled, not run on hardware
//
#include <hip/hip_runtime.h>
#include <hip/hip_bf16.h>

// ---------------------------------------------------------------------------
// QuantumNAT: pooled features (WMMA GEMM) -> 4-qubit circuit -> batchnorm
// MI455X / gfx1250, wave32. Memory-bound (~302 MB reads @ 23.3 TB/s ~ 13 us).
// ---------------------------------------------------------------------------

typedef __attribute__((ext_vector_type(16))) __bf16 v16bf;
typedef __attribute__((ext_vector_type(8)))  float  v8f;

#define BN_EPS 1e-5f
#define AST            584   // padded ushorts per staged image row (576 + 8), 16B-aligned
#define IMGS_PER_WAVE  16
#define WAVES          8
#define IMGS_PER_BLOCK 128   // WAVES * IMGS_PER_WAVE

union ABf { uint4 q[2]; v16bf v; };
union BBf { unsigned short h[16]; v16bf v; };

__device__ __forceinline__ unsigned short f2bf(float f) {
  union { float f; unsigned u; } v; v.f = f;
  unsigned u = v.u;
  u += 0x7FFFu + ((u >> 16) & 1u);   // round-to-nearest-even
  return (unsigned short)(u >> 16);
}

// ---------------- quantum gates (BIT is compile-time -> pure VGPR state) ----
template<int BIT>
__device__ __forceinline__ void gate_ry(float* sr, float* si, float th) {
  float s = __sinf(th * 0.5f), c = __cosf(th * 0.5f);
#pragma unroll
  for (int i = 0; i < 16; ++i) {
    if ((i & BIT) == 0) {
      const int j = i | BIT;
      float r0 = sr[i], i0 = si[i], r1 = sr[j], i1 = si[j];
      sr[i] = c * r0 - s * r1;  si[i] = c * i0 - s * i1;
      sr[j] = s * r0 + c * r1;  si[j] = s * i0 + c * i1;
    }
  }
}
template<int BIT>
__device__ __forceinline__ void gate_rz(float* sr, float* si, float th) {
  float s = __sinf(th * 0.5f), c = __cosf(th * 0.5f);
#pragma unroll
  for (int i = 0; i < 16; ++i) {
    float r = sr[i], im = si[i];
    if ((i & BIT) == 0) { sr[i] = r * c + im * s; si[i] = im * c - r * s; }
    else                { sr[i] = r * c - im * s; si[i] = im * c + r * s; }
  }
}
template<int BC, int BT>
__device__ __forceinline__ void gate_cnot(float* sr, float* si) {
#pragma unroll
  for (int i = 0; i < 16; ++i) {
    if ((i & BC) && !(i & BT)) {
      const int j = i | BT;
      float tr = sr[i], ti = si[i];
      sr[i] = sr[j]; si[i] = si[j];
      sr[j] = tr;    si[j] = ti;
    }
  }
}

// ---------------------------------------------------------------------------
// K1: stage -> WMMA pooling -> circuit -> raw z + per-block BN partials
// ---------------------------------------------------------------------------
__global__ __launch_bounds__(256) void k_qnat_main(
    const float* __restrict__ x,     // [B][1][28][28]
    const float* __restrict__ enc,   // [4][2]
    const float* __restrict__ vp,    // [2][4][3]
    float* __restrict__ raw,         // [B][4]
    float* __restrict__ partials)    // [nblk][8]  (sum z, sum z^2)
{
  extern __shared__ char smem[];
  unsigned short* aLDS  = (unsigned short*)smem;                        // 8*16*AST ushorts
  float* featLDS = (float*)(smem + (size_t)WAVES * IMGS_PER_WAVE * AST * 2); // 128*4 floats
  float* bnLDS   = featLDS + IMGS_PER_BLOCK * 4;                        // 8*8 floats

  const int tid  = threadIdx.x;
  const int w    = tid >> 5;
  const int lane = tid & 31;
  const int waveImg0 = blockIdx.x * IMGS_PER_BLOCK + w * IMGS_PER_WAVE;

  // L2-streaming hint for data a few blocks ahead (global_prefetch_b8)
  __builtin_prefetch(x + (size_t)(waveImg0 + 4 * IMGS_PER_BLOCK) * 784 + lane * 4, 0, 1);

  // ---- stage: 16 images x 24 rows x 24 cols, fp32 -> bf16 into LDS --------
  unsigned short* aw = aLDS + (size_t)w * IMGS_PER_WAVE * AST;
#pragma unroll 4
  for (int i = 0; i < 72; ++i) {            // 16*24*6 float4 / 32 lanes
    int flat = lane + 32 * i;               // [0, 2304)
    int m    = flat / 144;                  // image in tile
    int rem  = flat - m * 144;
    int r    = rem / 6;                     // row 0..23
    int c4   = rem - r * 6;                 // float4 column 0..5
    const float4 v4 = *(const float4*)(x + (size_t)(waveImg0 + m) * 784 + r * 28 + c4 * 4);
    unsigned p0 = (unsigned)f2bf(v4.x) | ((unsigned)f2bf(v4.y) << 16);
    unsigned p1 = (unsigned)f2bf(v4.z) | ((unsigned)f2bf(v4.w) << 16);
    *(uint2*)(aw + m * AST + r * 24 + c4 * 4) = make_uint2(p0, p1);
  }
  __syncthreads();

  // ---- pooling GEMM: C[m][n] = sum_k A[m,k] * B[k,n], B[k,n] = (k/144==n) --
  // A 16x32 bf16 layout: lane<16 row m holds k {0..7,16..23}; lane>=16 {8..15,24..31}
  // B 32x16 bf16 layout: same k order as A: lanes 0-15 hold K {0..7,16..23}, lanes 16-31 {8..15,24..31}
  const int mrow = lane & 15;
  const int kh   = lane >> 4;
  const int n144 = mrow * 144;
  v8f acc = {0.f, 0.f, 0.f, 0.f, 0.f, 0.f, 0.f, 0.f};
#pragma unroll
  for (int j = 0; j < 18; ++j) {            // 18 * 32 = 576 = full K
    ABf a;
    const unsigned short* ap = aw + mrow * AST + 32 * j + 8 * kh;
    a.q[0] = *(const uint4*)(ap);
    a.q[1] = *(const uint4*)(ap + 16);
    BBf b;
    const int kB = 32 * j + 8 * kh;
#pragma unroll
    for (int e = 0; e < 16; ++e)
      b.h[e] = ((unsigned)(kB + e + (e & 8) - n144) < 144u) ? (unsigned short)0x3F80 : (unsigned short)0;
    acc = __builtin_amdgcn_wmma_f32_16x16x32_bf16(false, a.v, false, b.v,
                                                  (short)0, acc, false, false);
  }
  // C layout: lane L holds column n=L&15, rows M = v + 8*(L>>4)
  if (mrow < 4) {
#pragma unroll
    for (int v = 0; v < 8; ++v)
      featLDS[(w * IMGS_PER_WAVE + v + 8 * kh) * 4 + mrow] = acc[v] * (1.0f / 144.0f);
  }
  __syncthreads();

  // ---- 4-qubit circuit, one image per lane (lanes 0..15 of each wave) -----
  float z[4]  = {0.f, 0.f, 0.f, 0.f};
  float z2[4] = {0.f, 0.f, 0.f, 0.f};
  if (lane < IMGS_PER_WAVE) {
    const int limg = w * IMGS_PER_WAVE + lane;
    const float f0 = featLDS[limg * 4 + 0];
    const float f1 = featLDS[limg * 4 + 1];
    const float f2 = featLDS[limg * 4 + 2];
    const float f3 = featLDS[limg * 4 + 3];

    float sr[16], si[16];
#pragma unroll
    for (int i = 0; i < 16; ++i) { sr[i] = 0.f; si[i] = 0.f; }
    sr[0] = 1.f;

    // encoding: RY(enc[q,0]*f_q), RZ(enc[q,1]*f_q); qubit q <-> bit 8>>q
    gate_ry<8>(sr, si, enc[0] * f0);  gate_rz<8>(sr, si, enc[1] * f0);
    gate_ry<4>(sr, si, enc[2] * f1);  gate_rz<4>(sr, si, enc[3] * f1);
    gate_ry<2>(sr, si, enc[4] * f2);  gate_rz<2>(sr, si, enc[5] * f2);
    gate_ry<1>(sr, si, enc[6] * f3);  gate_rz<1>(sr, si, enc[7] * f3);

#pragma unroll
    for (int L = 0; L < 2; ++L) {
      const float* v = vp + L * 12;
      gate_ry<8>(sr, si, v[0]);  gate_rz<8>(sr, si, v[1]);  gate_ry<8>(sr, si, v[2]);
      gate_ry<4>(sr, si, v[3]);  gate_rz<4>(sr, si, v[4]);  gate_ry<4>(sr, si, v[5]);
      gate_ry<2>(sr, si, v[6]);  gate_rz<2>(sr, si, v[7]);  gate_ry<2>(sr, si, v[8]);
      gate_ry<1>(sr, si, v[9]);  gate_rz<1>(sr, si, v[10]); gate_ry<1>(sr, si, v[11]);
      gate_cnot<8, 4>(sr, si);
      gate_cnot<4, 2>(sr, si);
      gate_cnot<2, 1>(sr, si);
      gate_cnot<1, 8>(sr, si);
    }

#pragma unroll
    for (int i = 0; i < 16; ++i) {
      float p = sr[i] * sr[i] + si[i] * si[i];
#pragma unroll
      for (int q = 0; q < 4; ++q)
        z[q] += ((i >> (3 - q)) & 1) ? -p : p;
    }
#pragma unroll
    for (int q = 0; q < 4; ++q) z2[q] = z[q] * z[q];

    const int gimg = waveImg0 + lane;
    *(float4*)(raw + (size_t)gimg * 4) = make_float4(z[0], z[1], z[2], z[3]);
  }

  // ---- deterministic BN partials: fixed-order shuffle + LDS tree ----------
#pragma unroll
  for (int off = 16; off >= 1; off >>= 1) {
#pragma unroll
    for (int q = 0; q < 4; ++q) {
      z[q]  += __shfl_xor(z[q],  off, 32);
      z2[q] += __shfl_xor(z2[q], off, 32);
    }
  }
  if (lane == 0) {
#pragma unroll
    for (int q = 0; q < 4; ++q) {
      bnLDS[w * 8 + q]     = z[q];
      bnLDS[w * 8 + 4 + q] = z2[q];
    }
  }
  __syncthreads();
  if (tid < 8) {
    float s = 0.f;
#pragma unroll
    for (int ww = 0; ww < WAVES; ++ww) s += bnLDS[ww * 8 + tid];
    partials[blockIdx.x * 8 + tid] = s;
  }
}

// ---------------------------------------------------------------------------
// K2: reduce per-block partials in fixed order -> scale/bias per qubit
// ---------------------------------------------------------------------------
__global__ void k_qnat_stats(const float* __restrict__ partials, int nblk,
                             const float* __restrict__ gamma,
                             const float* __restrict__ beta,
                             float* __restrict__ stats, float invB)
{
  const int q = threadIdx.x;
  if (q < 4) {
    float s = 0.f, s2 = 0.f;
    for (int b = 0; b < nblk; ++b) {
      s  += partials[b * 8 + q];
      s2 += partials[b * 8 + 4 + q];
    }
    float mean = s * invB;
    float var  = s2 * invB - mean * mean;   // biased variance (jnp.var default)
    float sc   = gamma[q] / sqrtf(var + BN_EPS);
    stats[q]     = sc;
    stats[4 + q] = beta[q] - mean * sc;
  }
}

// ---------------------------------------------------------------------------
// K3: streaming normalize: out = raw*scale + bias
// ---------------------------------------------------------------------------
__global__ __launch_bounds__(256) void k_qnat_norm(const float* __restrict__ raw,
                                                   const float* __restrict__ stats,
                                                   float* __restrict__ out, int B)
{
  const int i = blockIdx.x * blockDim.x + threadIdx.x;
  if (i < B) {
    float4 r  = *(const float4*)(raw + (size_t)i * 4);
    float4 sc = *(const float4*)(stats);
    float4 bi = *(const float4*)(stats + 4);
    *(float4*)(out + (size_t)i * 4) =
        make_float4(r.x * sc.x + bi.x, r.y * sc.y + bi.y,
                    r.z * sc.z + bi.z, r.w * sc.w + bi.w);
  }
}

// ---------------------------------------------------------------------------
extern "C" void kernel_launch(void* const* d_in, const int* in_sizes, int n_in,
                              void* d_out, int out_size, void* d_ws, size_t ws_size,
                              hipStream_t stream)
{
  const float* x     = (const float*)d_in[0];
  const float* enc   = (const float*)d_in[1];
  const float* vp    = (const float*)d_in[2];
  const float* gamma = (const float*)d_in[3];
  const float* beta  = (const float*)d_in[4];

  const int B    = in_sizes[0] / 784;      // 131072
  const int nblk = B / IMGS_PER_BLOCK;     // 1024

  float* raw      = (float*)d_ws;          // [B][4]
  float* partials = raw + (size_t)B * 4;   // [nblk][8]
  float* stats    = partials + (size_t)nblk * 8; // [8]

  const size_t smem = (size_t)WAVES * IMGS_PER_WAVE * AST * 2   // bf16 A stage
                    + (size_t)IMGS_PER_BLOCK * 4 * sizeof(float) // features
                    + (size_t)WAVES * 8 * sizeof(float);         // BN scratch

  k_qnat_main <<<nblk, 256, smem, stream>>>(x, enc, vp, raw, partials);
  k_qnat_stats<<<1, 32, 0, stream>>>(partials, nblk, gamma, beta, stats, 1.0f / (float)B);
  k_qnat_norm <<<(B + 255) / 256, 256, 0, stream>>>(raw, stats, (float*)d_out, B);
}
